// GEN_15247133900994
// MI455X (gfx1250) — matrix operand order repaired and checked by reading, not compiled, not run on hardware
//
#include <hip/hip_runtime.h>
#include <hip/hip_bf16.h>
#include <math.h>

#define SIDE      32
#define N_NODES   1024
#define WIDTH     128
#define IN_DIM    128
#define T_DIM     2
#define BATCH     8
#define MSG_STEPS 32
#define LN_EPS    1e-5f
#define MAXD      8
#define ROWS      (BATCH * N_NODES)   // 8192

typedef __attribute__((ext_vector_type(16))) _Float16 v16h;
typedef __attribute__((ext_vector_type(8)))  _Float16 v8h;
typedef __attribute__((ext_vector_type(8)))  float    v8f;

// ---------------- encoder MLP: enc[8][128] ----------------
__global__ void k_encoder(const float* __restrict__ X1,
                          const float* __restrict__ We1, const float* __restrict__ be1,
                          const float* __restrict__ We2, const float* __restrict__ be2,
                          const float* __restrict__ We3, const float* __restrict__ be3,
                          float* __restrict__ enc) {
  __shared__ float h1[BATCH][WIDTH];
  __shared__ float h2[BATCH][WIDTH];
  const int t = threadIdx.x;
  for (int i = t; i < BATCH * WIDTH; i += blockDim.x) {
    int b = i >> 7, c = i & 127;
    float acc = be1[c];
    for (int k = 0; k < IN_DIM; ++k)
      acc = fmaf(X1[b * (IN_DIM + T_DIM) + k], We1[k * WIDTH + c], acc);
    h1[b][c] = fmaxf(acc, 0.f);
  }
  __syncthreads();
  for (int i = t; i < BATCH * WIDTH; i += blockDim.x) {
    int b = i >> 7, c = i & 127;
    float acc = be2[c];
    for (int k = 0; k < WIDTH; ++k) acc = fmaf(h1[b][k], We2[k * WIDTH + c], acc);
    h2[b][c] = fmaxf(acc, 0.f);
  }
  __syncthreads();
  for (int i = t; i < BATCH * WIDTH; i += blockDim.x) {
    int b = i >> 7, c = i & 127;
    float acc = be3[c];
    for (int k = 0; k < WIDTH; ++k) acc = fmaf(h2[b][k], We3[k * WIDTH + c], acc);
    enc[i] = acc;
  }
}

// ---------------- w0 = softmax(-||pos||) over nodes (1 block, 1024 thr) ----------------
__global__ void k_w0(const float* __restrict__ pos, float* __restrict__ w0) {
  __shared__ float red[32];
  const int n = threadIdx.x;
  float px = pos[2 * n], py = pos[2 * n + 1];
  float neg = -sqrtf(px * px + py * py);
  float m = neg;
  for (int off = 16; off; off >>= 1) m = fmaxf(m, __shfl_xor(m, off, 32));
  if ((n & 31) == 0) red[n >> 5] = m;
  __syncthreads();
  if (n < 32) {
    float mm = red[n];
    for (int off = 16; off; off >>= 1) mm = fmaxf(mm, __shfl_xor(mm, off, 32));
    if (n == 0) red[0] = mm;
  }
  __syncthreads();
  const float M = red[0];
  __syncthreads();
  float e = __expf(neg - M);
  float s = e;
  for (int off = 16; off; off >>= 1) s += __shfl_xor(s, off, 32);
  if ((n & 31) == 0) red[n >> 5] = s;
  __syncthreads();
  if (n < 32) {
    float ss = red[n];
    for (int off = 16; off; off >>= 1) ss += __shfl_xor(ss, off, 32);
    if (n == 0) red[0] = ss;
  }
  __syncthreads();
  w0[n] = e / red[0];
}

// ---------------- x init: x[b,n,c] = w0[n]*enc[b,c] (fp32 + f16 copies) ----------------
__global__ void k_xinit(const float* __restrict__ w0, const float* __restrict__ enc,
                        float* __restrict__ X32, _Float16* __restrict__ X16) {
  const int bn = blockIdx.x, c = threadIdx.x;
  const int b = bn >> 10, node = bn & (N_NODES - 1);
  float v = w0[node] * enc[b * WIDTH + c];
  X32[(size_t)bn * WIDTH + c] = v;
  X16[(size_t)bn * WIDTH + c] = (_Float16)v;
}

// ---------------- graph setup ----------------
__global__ void k_zero_i32(int* __restrict__ p, int n) {
  int i = blockIdx.x * blockDim.x + threadIdx.x;
  if (i < n) p[i] = 0;
}
__global__ void k_build(const int* __restrict__ ei, int E,
                        int* __restrict__ cnt, int* __restrict__ nbr) {
  int e = blockIdx.x * blockDim.x + threadIdx.x;
  if (e >= E) return;
  int c = ei[E + e];                    // col
  int i = atomicAdd(&cnt[c], 1);
  if (i < MAXD) nbr[c * MAXD + i] = e;
}
__global__ void k_norms(const int* __restrict__ cnt,
                        float* __restrict__ dinv, float* __restrict__ snorm) {
  int n = blockIdx.x * blockDim.x + threadIdx.x;
  if (n >= N_NODES) return;
  float di = rsqrtf(1.0f + (float)cnt[n]);   // deg = 1 + in-count
  dinv[n] = di;
  snorm[n] = di * di;
}
__global__ void k_enorm(const int* __restrict__ ei, int E,
                        const float* __restrict__ dinv, float* __restrict__ enorm) {
  int e = blockIdx.x * blockDim.x + threadIdx.x;
  if (e >= E) return;
  enorm[e] = dinv[ei[e]] * dinv[ei[E + e]];
}

// ---------------- step-invariant precompute ----------------
// posW[n][c] = pos[n]·Wg[:2, c]   (folded into WMMA accumulator init)
__global__ void k_posw(const float* __restrict__ pos, const float* __restrict__ Wg,
                       float* __restrict__ posW) {
  const int n = blockIdx.x, c = threadIdx.x;
  posW[n * WIDTH + c] = pos[2 * n] * Wg[c] + pos[2 * n + 1] * Wg[WIDTH + c];
}
// WgT16[n][k] = (f16) Wg[k+2][n]  (N-major B operand)
__global__ void k_wgt(const float* __restrict__ Wg, _Float16* __restrict__ WgT) {
  const int n = blockIdx.x, k = threadIdx.x;
  WgT[n * WIDTH + k] = (_Float16)Wg[(k + T_DIM) * WIDTH + n];
}

// ---------------- the hot GEMM: XW = X16 @ WgT^T + posW, via WMMA ----------------
__global__ void __launch_bounds__(256)
k_gemm(const _Float16* __restrict__ X16, const _Float16* __restrict__ WgT,
       const float* __restrict__ posW, float* __restrict__ XW) {
  const int lane  = threadIdx.x & 31;
  const int wv    = threadIdx.x >> 5;       // n-tile index (8 waves -> 128 cols)
  const int mrow  = blockIdx.x << 4;        // 16 rows per block
  const int ncol0 = wv << 4;
  const int idx16 = lane & 15;
  const int khalf = lane >> 4;

  const _Float16* arow = X16 + (size_t)(mrow + idx16) * WIDTH;   // A row = row m
  const _Float16* brow = WgT + (size_t)(ncol0 + idx16) * WIDTH;  // B "row" = col n

  // C/D layout: VGPR r, lane L -> (m = r + 8*(L>>4), n = L&15)
  v8f c;
  const int mc = khalf << 3;
  const int nc = idx16;
#pragma unroll
  for (int r = 0; r < 8; ++r) {
    int node = (mrow + mc + r) & (N_NODES - 1);
    c[r] = posW[node * WIDTH + ncol0 + nc];
  }

#pragma unroll
  for (int kc = 0; kc < 4; ++kc) {   // K = 128 in chunks of 32
    const int k0 = kc << 5;
    union { v16h v; v8h h[2]; } a, b;
    // A 16x32 f16: V0-3 K = khalf*8 + [0..7], V4-7 K = 16 + khalf*8 + [0..7]
    a.h[0] = *(const v8h*)(arow + k0 + (khalf << 3));
    a.h[1] = *(const v8h*)(arow + k0 + 16 + (khalf << 3));
    // B 32x16 f16: V0-3 K = khalf*8 + [0..7], V4-7 K = 16 + khalf*8 + [0..7]
    b.h[0] = *(const v8h*)(brow + k0 + (khalf << 3));
    b.h[1] = *(const v8h*)(brow + k0 + 16 + (khalf << 3));
    c = __builtin_amdgcn_wmma_f32_16x16x32_f16(false, a.v, false, b.v,
                                               (short)0, c, false, false);
  }

#pragma unroll
  for (int r = 0; r < 8; ++r) {
    int gm = mrow + mc + r;
    XW[(size_t)gm * WIDTH + ncol0 + nc] = c[r];
  }
}

// ---------------- gather messages + residual + LayerNorm (fused) ----------------
__global__ void k_msg_ln(const float* __restrict__ XW,
                         const int* __restrict__ cnt, const int* __restrict__ nbr,
                         const int* __restrict__ ei, int E,
                         const float* __restrict__ enorm, const float* __restrict__ snorm,
                         const float* __restrict__ bg,
                         const float* __restrict__ gamma, const float* __restrict__ beta,
                         float* __restrict__ X32, _Float16* __restrict__ X16) {
  __shared__ float redS[4], redQ[4];
  const int bn = blockIdx.x;
  const int node = bn & (N_NODES - 1);
  const int b = bn >> 10;
  const int c = threadIdx.x;

  float v = snorm[node] * XW[(size_t)bn * WIDTH + c] + bg[c];
  const int deg_in = min(cnt[node], MAXD);
  for (int i = 0; i < deg_in; ++i) {
    int e = nbr[node * MAXD + i];
    int src = ei[e];                                     // row[e]
    v += enorm[e] * XW[(size_t)((b << 10) + src) * WIDTH + c];
  }
  float xn = X32[(size_t)bn * WIDTH + c] + v;

  float s = xn, q = xn * xn;
  for (int off = 16; off; off >>= 1) {
    s += __shfl_xor(s, off, 32);
    q += __shfl_xor(q, off, 32);
  }
  if ((c & 31) == 0) { redS[c >> 5] = s; redQ[c >> 5] = q; }
  __syncthreads();
  float S = redS[0] + redS[1] + redS[2] + redS[3];
  float Q = redQ[0] + redQ[1] + redQ[2] + redQ[3];
  float mu = S * (1.f / WIDTH);
  float var = Q * (1.f / WIDTH) - mu * mu;
  float o = (xn - mu) * rsqrtf(var + LN_EPS) * gamma[c] + beta[c];
  X32[(size_t)bn * WIDTH + c] = o;
  X16[(size_t)bn * WIDTH + c] = (_Float16)o;
}

// ---------------- decoder pooling: hidden[b] = softmax(-dist) · x ----------------
__global__ void k_pool(const float* __restrict__ pos, const float* __restrict__ X1,
                       const float* __restrict__ X32, float* __restrict__ hidden) {
  __shared__ float wt[N_NODES];
  __shared__ float red[8];
  const int b = blockIdx.x;
  const int tid = threadIdx.x;
  const float t0 = X1[b * (IN_DIM + T_DIM) + IN_DIM];
  const float t1 = X1[b * (IN_DIM + T_DIM) + IN_DIM + 1];

  float m = -3.4e38f;
  for (int n = tid; n < N_NODES; n += 256) {
    float dx = pos[2 * n] - t0, dy = pos[2 * n + 1] - t1;
    float nd = -sqrtf(dx * dx + dy * dy);
    wt[n] = nd;
    m = fmaxf(m, nd);
  }
  for (int off = 16; off; off >>= 1) m = fmaxf(m, __shfl_xor(m, off, 32));
  if ((tid & 31) == 0) red[tid >> 5] = m;
  __syncthreads();
  if (tid == 0) { float M = red[0]; for (int i = 1; i < 8; ++i) M = fmaxf(M, red[i]); red[0] = M; }
  __syncthreads();
  const float M = red[0];
  __syncthreads();
  float ssum = 0.f;
  for (int n = tid; n < N_NODES; n += 256) {
    float e = __expf(wt[n] - M);
    wt[n] = e;
    ssum += e;
  }
  for (int off = 16; off; off >>= 1) ssum += __shfl_xor(ssum, off, 32);
  if ((tid & 31) == 0) red[tid >> 5] = ssum;
  __syncthreads();
  if (tid == 0) { float S = 0.f; for (int i = 0; i < 8; ++i) S += red[i]; red[0] = S; }
  __syncthreads();
  const float invS = 1.f / red[0];
  for (int w = tid; w < WIDTH; w += 256) {
    float acc = 0.f;
    for (int n = 0; n < N_NODES; ++n)
      acc = fmaf(wt[n], X32[(size_t)((b << 10) + n) * WIDTH + w], acc);
    hidden[b * (WIDTH + T_DIM) + w] = acc * invS;
  }
  if (tid == 0) {
    hidden[b * (WIDTH + T_DIM) + WIDTH] = t0;
    hidden[b * (WIDTH + T_DIM) + WIDTH + 1] = t1;
  }
}

// ---------------- decoder MLP -> y[8] ----------------
__global__ void k_decoder(const float* __restrict__ hidden,
                          const float* __restrict__ Wd1, const float* __restrict__ bd1,
                          const float* __restrict__ Wd2, const float* __restrict__ bd2,
                          const float* __restrict__ Wd3, const float* __restrict__ bd3,
                          float* __restrict__ y) {
  __shared__ float h1[BATCH][WIDTH];
  __shared__ float h2[BATCH][WIDTH];
  const int t = threadIdx.x;
  for (int i = t; i < BATCH * WIDTH; i += blockDim.x) {
    int b = i >> 7, c = i & 127;
    float acc = bd1[c];
    for (int k = 0; k < WIDTH + T_DIM; ++k)
      acc = fmaf(hidden[b * (WIDTH + T_DIM) + k], Wd1[k * WIDTH + c], acc);
    h1[b][c] = fmaxf(acc, 0.f);
  }
  __syncthreads();
  for (int i = t; i < BATCH * WIDTH; i += blockDim.x) {
    int b = i >> 7, c = i & 127;
    float acc = bd2[c];
    for (int k = 0; k < WIDTH; ++k) acc = fmaf(h1[b][k], Wd2[k * WIDTH + c], acc);
    h2[b][c] = fmaxf(acc, 0.f);
  }
  __syncthreads();
  if (t < BATCH) {
    float acc = bd3[0];
    for (int k = 0; k < WIDTH; ++k) acc = fmaf(h2[t][k], Wd3[k], acc);
    y[t] = acc;
  }
}

extern "C" void kernel_launch(void* const* d_in, const int* in_sizes, int n_in,
                              void* d_out, int out_size, void* d_ws, size_t ws_size,
                              hipStream_t stream) {
  const float* X1    = (const float*)d_in[0];
  const float* pos   = (const float*)d_in[1];
  const int*   ei    = (const int*)d_in[2];
  const float* We1   = (const float*)d_in[3];
  const float* be1   = (const float*)d_in[4];
  const float* We2   = (const float*)d_in[5];
  const float* be2   = (const float*)d_in[6];
  const float* We3   = (const float*)d_in[7];
  const float* be3   = (const float*)d_in[8];
  const float* Wg    = (const float*)d_in[9];
  const float* bg    = (const float*)d_in[10];
  const float* gamma = (const float*)d_in[11];
  const float* beta  = (const float*)d_in[12];
  const float* Wd1   = (const float*)d_in[13];
  const float* bd1   = (const float*)d_in[14];
  const float* Wd2   = (const float*)d_in[15];
  const float* bd2   = (const float*)d_in[16];
  const float* Wd3   = (const float*)d_in[17];
  const float* bd3   = (const float*)d_in[18];
  float* y = (float*)d_out;
  const int E = in_sizes[2] / 2;

  char* w = (char*)d_ws;
  size_t off = 0;
  auto carve = [&](size_t bytes) -> void* {
    void* p = w + off;
    off = (off + bytes + 255) & ~(size_t)255;
    return p;
  };
  float*    enc   = (float*)   carve((size_t)BATCH * WIDTH * 4);
  float*    w0    = (float*)   carve((size_t)N_NODES * 4);
  float*    X32   = (float*)   carve((size_t)ROWS * WIDTH * 4);
  _Float16* X16   = (_Float16*)carve((size_t)ROWS * WIDTH * 2);
  float*    XW    = (float*)   carve((size_t)ROWS * WIDTH * 4);
  float*    posW  = (float*)   carve((size_t)N_NODES * WIDTH * 4);
  _Float16* WgT   = (_Float16*)carve((size_t)WIDTH * WIDTH * 2);
  int*      cnt   = (int*)     carve((size_t)N_NODES * 4);
  int*      nbr   = (int*)     carve((size_t)N_NODES * MAXD * 4);
  float*    dinv  = (float*)   carve((size_t)N_NODES * 4);
  float*    snorm = (float*)   carve((size_t)N_NODES * 4);
  float*    enorm = (float*)   carve((size_t)E * 4);
  float*    hid   = (float*)   carve((size_t)BATCH * (WIDTH + T_DIM) * 4);

  // one-time (per call) setup
  k_encoder<<<1, 256, 0, stream>>>(X1, We1, be1, We2, be2, We3, be3, enc);
  k_w0<<<1, 1024, 0, stream>>>(pos, w0);
  k_xinit<<<ROWS, WIDTH, 0, stream>>>(w0, enc, X32, X16);
  k_zero_i32<<<(N_NODES + 255) / 256, 256, 0, stream>>>(cnt, N_NODES);
  k_build<<<(E + 255) / 256, 256, 0, stream>>>(ei, E, cnt, nbr);
  k_norms<<<(N_NODES + 255) / 256, 256, 0, stream>>>(cnt, dinv, snorm);
  k_enorm<<<(E + 255) / 256, 256, 0, stream>>>(ei, E, dinv, enorm);
  k_posw<<<N_NODES, WIDTH, 0, stream>>>(pos, Wg, posW);
  k_wgt<<<WIDTH, WIDTH, 0, stream>>>(Wg, WgT);

  // message-passing loop: WMMA GEMM + fused gather/LN
  for (int step = 0; step < MSG_STEPS; ++step) {
    k_gemm<<<ROWS / 16, 256, 0, stream>>>(X16, WgT, posW, XW);
    k_msg_ln<<<ROWS, WIDTH, 0, stream>>>(XW, cnt, nbr, ei, E, enorm, snorm,
                                         bg, gamma, beta, X32, X16);
  }

  // decoder
  k_pool<<<BATCH, 256, 0, stream>>>(pos, X1, X32, hid);
  k_decoder<<<1, 256, 0, stream>>>(hid, Wd1, bd1, Wd2, bd2, Wd3, bd3, y);
}
